// Model_hier_LSTM_10247791968774
// MI455X (gfx1250) — compile-verified, matrix operand order checked by reading, not run on hardware
//
#include <hip/hip_runtime.h>
#include <hip/hip_bf16.h>

// ---------------------------------------------------------------------------
// Types for CDNA5 WMMA (wave32): v_wmma_f32_16x16x32_bf16
// ---------------------------------------------------------------------------
typedef __attribute__((ext_vector_type(16))) __bf16 v16bf;
typedef __attribute__((ext_vector_type(8)))  float  v8f;

static __device__ __forceinline__ unsigned short f2bf(float f) {
    unsigned u = __float_as_uint(f);
    u += 0x7FFFu + ((u >> 16) & 1u);          // round-to-nearest-even
    return (unsigned short)(u >> 16);
}
static __device__ __forceinline__ float bf2f(unsigned short h) {
    return __uint_as_float(((unsigned)h) << 16);
}
static __device__ __forceinline__ float sigf(float x) {
    return 1.f / (1.f + __expf(-x));
}

// ---------------------------------------------------------------------------
// Generic tiled GEMM:  out = act( A[M,K] @ W[N,K]^T + bias )
//   A: fp32 (aBf16==0) or bf16 (aBf16==1), row stride lda (elements)
//   W: bf16, row stride ldw (elements), rows are output features
//   outF (fp32) and/or outB (bf16), row stride ldc
//   act: 0 = none, 1 = ELU
// Block: 256 threads = 8 waves; tile 128(M) x 64(N) x 32(K);
// each wave computes a 32x32 output patch = 2x2 WMMA fragments.
// Fast path (full tiles): double-buffered LDS fed by
// global_load_async_to_lds_b128 (ASYNCcnt), overlapped with WMMA.
// ---------------------------------------------------------------------------
#define BM 128
#define BN 64
#define BK 32
#define LDSS 40   // LDS row stride in bf16 elems (80B -> bank-conflict free)
#define SA_BUF (BM * LDSS)
#define SB_BUF (BN * LDSS)

__global__ __launch_bounds__(256) void gemm_bf16_kernel(
    const void* __restrict__ Aptr, int aBf16, int lda,
    const unsigned short* __restrict__ W, int ldw,
    const float* __restrict__ bias, int act,
    float* __restrict__ outF, unsigned short* __restrict__ outB, int ldc,
    int M, int N, int K)
{
    __shared__ unsigned short sA[2 * SA_BUF];
    __shared__ unsigned short sB[2 * SB_BUF];

    const int tid    = threadIdx.x;
    const int blockM = blockIdx.y * BM;
    const int blockN = blockIdx.x * BN;

    const int wave  = tid >> 5;
    const int lane  = tid & 31;
    const int wm    = (wave & 3) * 32;   // 4 waves along M
    const int wn    = (wave >> 2) * 32;  // 2 waves along N
    const int lrow  = lane & 15;
    const int lhalf = lane >> 4;

    v8f acc[2][2];
    for (int i = 0; i < 2; ++i)
        for (int j = 0; j < 2; ++j)
            for (int e = 0; e < 8; ++e) acc[i][j][e] = 0.f;

    const float*          Af = (const float*)Aptr;
    const unsigned short* Ab = (const unsigned short*)Aptr;

    const bool edge = ((blockM + BM) > M) || ((blockN + BN) > N);
    const int  Kvec = K & ~(BK - 1);

    // ---- guard-free tile staging (requires full M/N tile, k0+BK <= K) ----
    auto loadFast = [&](int k0, int buf) {
        if (aBf16) {
            for (int i = 0; i < 2; ++i) {
                int g = tid + i * 256;                  // 0..511 chunks of 8
                int r = g >> 2;
                int c = (g & 3) * 8;
                const unsigned short* gp = Ab + (size_t)(blockM + r) * lda + (k0 + c);
                unsigned dst = (unsigned)(unsigned long long)&sA[buf * SA_BUF + r * LDSS + c];
                asm volatile("global_load_async_to_lds_b128 %0, %1, off"
                             :: "v"(dst), "v"(gp) : "memory");
            }
        } else {
            for (int i = 0; i < 4; ++i) {
                int g = tid + i * 256;                  // 0..1023 chunks of 4
                int r = g >> 3;
                int c = (g & 7) * 4;
                float4 v = *(const float4*)(Af + (size_t)(blockM + r) * lda + (k0 + c));
                ushort4 b4;
                b4.x = f2bf(v.x); b4.y = f2bf(v.y);
                b4.z = f2bf(v.z); b4.w = f2bf(v.w);
                *(ushort4*)&sA[buf * SA_BUF + r * LDSS + c] = b4;
            }
        }
        {
            int r = tid >> 2;
            int c = (tid & 3) * 8;
            const unsigned short* gp = W + (size_t)(blockN + r) * ldw + (k0 + c);
            unsigned dst = (unsigned)(unsigned long long)&sB[buf * SB_BUF + r * LDSS + c];
            asm volatile("global_load_async_to_lds_b128 %0, %1, off"
                         :: "v"(dst), "v"(gp) : "memory");
        }
    };

    // ---- fully guarded tile staging (edge blocks / K tail) ----
    auto loadSlow = [&](int k0, int buf) {
        if (aBf16) {
            for (int i = 0; i < 2; ++i) {
                int g = tid + i * 256;
                int r = g >> 2;
                int c = (g & 3) * 8;
                int grow = blockM + r;
                int k = k0 + c;
                if (grow < M && (k + 7) < K) {
                    *(uint4*)&sA[buf * SA_BUF + r * LDSS + c] =
                        *(const uint4*)(Ab + (size_t)grow * lda + k);
                } else {
                    for (int e = 0; e < 8; ++e) {
                        int kk = k + e;
                        sA[buf * SA_BUF + r * LDSS + c + e] = (grow < M && kk < K)
                            ? Ab[(size_t)grow * lda + kk] : (unsigned short)0;
                    }
                }
            }
        } else {
            for (int i = 0; i < 4; ++i) {
                int g = tid + i * 256;
                int r = g >> 3;
                int c = (g & 7) * 4;
                int grow = blockM + r;
                int k = k0 + c;
                float4 v = make_float4(0.f, 0.f, 0.f, 0.f);
                if (grow < M) {
                    if ((k + 3) < K) {
                        v = *(const float4*)(Af + (size_t)grow * lda + k);
                    } else {
                        v.x = (k + 0 < K) ? Af[(size_t)grow * lda + k + 0] : 0.f;
                        v.y = (k + 1 < K) ? Af[(size_t)grow * lda + k + 1] : 0.f;
                        v.z = (k + 2 < K) ? Af[(size_t)grow * lda + k + 2] : 0.f;
                        v.w = (k + 3 < K) ? Af[(size_t)grow * lda + k + 3] : 0.f;
                    }
                }
                ushort4 b4;
                b4.x = f2bf(v.x); b4.y = f2bf(v.y);
                b4.z = f2bf(v.z); b4.w = f2bf(v.w);
                *(ushort4*)&sA[buf * SA_BUF + r * LDSS + c] = b4;
            }
        }
        {
            int r = tid >> 2;
            int c = (tid & 3) * 8;
            int gn = blockN + r;
            int k = k0 + c;
            if (gn < N && (k + 7) < K) {
                *(uint4*)&sB[buf * SB_BUF + r * LDSS + c] =
                    *(const uint4*)(W + (size_t)gn * ldw + k);
            } else {
                for (int e = 0; e < 8; ++e) {
                    int kk = k + e;
                    sB[buf * SB_BUF + r * LDSS + c + e] = (gn < N && kk < K)
                        ? W[(size_t)gn * ldw + kk] : (unsigned short)0;
                }
            }
        }
    };

    // ---- fragment load + 4x WMMA for one staged 32-K slab ----
    // A/B 16-bit lane layout: lane<16: row L, K {0..7,16..23};
    //                         lane>=16: row L-16, K {8..15,24..31}
    auto mma = [&](int buf) {
        union FragU { uint4 u[2]; v16bf v; };
        FragU a[2], b[2];
        for (int fm = 0; fm < 2; ++fm) {
            int base = buf * SA_BUF + (wm + fm * 16 + lrow) * LDSS + lhalf * 8;
            a[fm].u[0] = *(const uint4*)&sA[base];
            a[fm].u[1] = *(const uint4*)&sA[base + 16];
        }
        for (int fn = 0; fn < 2; ++fn) {
            int base = buf * SB_BUF + (wn + fn * 16 + lrow) * LDSS + lhalf * 8;
            b[fn].u[0] = *(const uint4*)&sB[base];
            b[fn].u[1] = *(const uint4*)&sB[base + 16];
        }
        for (int fm = 0; fm < 2; ++fm)
            for (int fn = 0; fn < 2; ++fn)
                acc[fm][fn] = __builtin_amdgcn_wmma_f32_16x16x32_bf16(
                    false, a[fm].v, false, b[fn].v,
                    (short)0, acc[fm][fn], false, false);
    };

    if (!edge) {
        // pipelined: prefetch next slab (async -> other LDS buffer) while
        // current slab feeds the matrix cores
        if (Kvec > 0) loadFast(0, 0);
        int buf = 0;
        for (int k0 = 0; k0 < Kvec; k0 += BK) {
            bool hasNext = (k0 + BK) < Kvec;
            if (hasNext) loadFast(k0 + BK, buf ^ 1);
            // in-order ASYNCcnt: <=3 (or <=1 fp32-A) pending == current done
            if (hasNext) {
                if (aBf16) asm volatile("s_wait_asynccnt 0x3" ::: "memory");
                else       asm volatile("s_wait_asynccnt 0x1" ::: "memory");
            } else {
                asm volatile("s_wait_asynccnt 0x0" ::: "memory");
            }
            __syncthreads();
            mma(buf);
            __syncthreads();
            buf ^= 1;
        }
        if (Kvec < K) {                       // ragged K tail, once
            loadSlow(Kvec, 0);
            __syncthreads();
            mma(0);
            __syncthreads();
        }
    } else {                                  // ragged M/N block: cold path
        for (int k0 = 0; k0 < K; k0 += BK) {
            loadSlow(k0, 0);
            __syncthreads();
            mma(0);
            __syncthreads();
        }
    }

    // ---- epilogue: C/D layout lanes 0-15 -> M=i, lanes 16-31 -> M=i+8 ----
    for (int fm = 0; fm < 2; ++fm) {
        for (int fn = 0; fn < 2; ++fn) {
            int col = blockN + wn + fn * 16 + lrow;
            if (col >= N) continue;
            float bv = bias ? bias[col] : 0.f;
            for (int i = 0; i < 8; ++i) {
                int row = blockM + wm + fm * 16 + lhalf * 8 + i;
                if (row >= M) continue;
                float v = acc[fm][fn][i] + bv;
                if (act == 1) v = (v > 0.f) ? v : expm1f(v);   // ELU
                if (outF) outF[(size_t)row * ldc + col] = v;
                if (outB) outB[(size_t)row * ldc + col] = f2bf(v);
            }
        }
    }
}

// ---------------------------------------------------------------------------
// Weight conversion: fp32 (srcld stride, srcoff column offset) -> bf16 padded
// ---------------------------------------------------------------------------
__global__ void cvt_pad_kernel(const float* __restrict__ src, int srcld, int srcoff,
                               unsigned short* __restrict__ dst, int N, int K, int ldw)
{
    int idx = blockIdx.x * blockDim.x + threadIdx.x;
    if (idx >= N * ldw) return;
    int n = idx / ldw, k = idx - n * ldw;
    float v = (k < K) ? src[(size_t)n * srcld + srcoff + k] : 0.f;
    dst[idx] = f2bf(v);
}

// ---------------------------------------------------------------------------
// LSTM gate update:
//   g[q] = G[s, q*H+j] + Gx[gxRow, q*H+j] + bih[q*H+j] + bhh[q*H+j]
//   c = sig(f)*c + sig(i)*tanh(g);  h = sig(o)*tanh(c)
//   gxRow = (s/gxDiv)*gxOuter + (s%gxDiv)*gxInner + gxOff
// ---------------------------------------------------------------------------
__global__ void lstm_gate_kernel(const float* __restrict__ G,
                                 const unsigned short* __restrict__ Gx,
                                 int gxDiv, int gxOuter, int gxInner, int gxOff,
                                 const float* __restrict__ bih,
                                 const float* __restrict__ bhh,
                                 float* __restrict__ c,
                                 unsigned short* __restrict__ h, int ldh,
                                 int S, int H)
{
    int idx = blockIdx.x * blockDim.x + threadIdx.x;
    if (idx >= S * H) return;
    int s = idx / H, j = idx - s * H;
    size_t gxRow = (size_t)(s / gxDiv) * gxOuter + (size_t)(s % gxDiv) * gxInner + gxOff;
    float g[4];
    for (int q = 0; q < 4; ++q) {
        int col = q * H + j;
        g[q] = G[(size_t)s * (4 * H) + col]
             + bf2f(Gx[gxRow * (4 * H) + col])
             + bih[col] + bhh[col];
    }
    float gi = sigf(g[0]);
    float gf = sigf(g[1]);
    float gg = tanhf(g[2]);
    float go = sigf(g[3]);
    float cc = gf * c[idx] + gi * gg;
    c[idx] = cc;
    h[(size_t)s * ldh + j] = f2bf(go * tanhf(cc));
}

// ---------------------------------------------------------------------------
// Final: out[b] = relu( dot(h2[b,:], Wp) + bp );  one wave32 per row
// ---------------------------------------------------------------------------
__global__ void final_dot_kernel(const float* __restrict__ h2, int ldh,
                                 const float* __restrict__ Wp,
                                 const float* __restrict__ bp,
                                 float* __restrict__ out, int Bn, int H)
{
    int gwave = (blockIdx.x * blockDim.x + threadIdx.x) >> 5;
    int lane  = threadIdx.x & 31;
    if (gwave >= Bn) return;
    float sum = 0.f;
    for (int j = lane; j < H; j += 32)
        sum += h2[(size_t)gwave * ldh + j] * Wp[j];
    for (int o = 16; o; o >>= 1) sum += __shfl_down(sum, o, 32);
    if (lane == 0) {
        float v = sum + bp[0];
        out[gwave] = v > 0.f ? v : 0.f;
    }
}

// ---------------------------------------------------------------------------
// Host orchestration
// ---------------------------------------------------------------------------
extern "C" void kernel_launch(void* const* d_in, const int* in_sizes, int n_in,
                              void* d_out, int out_size, void* d_ws, size_t ws_size,
                              hipStream_t stream)
{
    (void)in_sizes; (void)n_in; (void)out_size; (void)ws_size;

    const float* tensor    = (const float*)d_in[0];
    const float* W_e1      = (const float*)d_in[1];
    const float* W_e2      = (const float*)d_in[2];
    const float* W_e3      = (const float*)d_in[3];
    const float* W_h1      = (const float*)d_in[4];
    const float* W_h2      = (const float*)d_in[5];
    const float* W_c1      = (const float*)d_in[6];
    const float* W_c2      = (const float*)d_in[7];
    const float* W_p       = (const float*)d_in[8];
    const float* b_p       = (const float*)d_in[9];
    const float* comps_Wih = (const float*)d_in[10];
    const float* comps_Whh = (const float*)d_in[11];
    const float* comps_bih = (const float*)d_in[12];
    const float* comps_bhh = (const float*)d_in[13];
    const float* nodes_Wih = (const float*)d_in[14];
    const float* nodes_Whh = (const float*)d_in[15];
    const float* nodes_bih = (const float*)d_in[16];
    const float* nodes_bhh = (const float*)d_in[17];
    float* out = (float*)d_out;

    const int Bq = 2048, NC = 64, IN = 228, H = 180;
    const int M0 = Bq * NC;              // 131072

    // ---- workspace arena ----
    char* ws = (char*)d_ws;
    size_t off = 0;
    auto take = [&](size_t bytes) -> char* {
        char* p = ws + off;
        off += (bytes + 255) & ~(size_t)255;
        return p;
    };
    size_t regA_sz = (size_t)M0 * 720 * 2;                 // max(enc1 600, Gx 720) bf16
    char* regA   = take(regA_sz);                          // enc1 out, then Gx(comps)
    char* regB   = take((size_t)M0 * 352 * 2);             // enc2 out, then G step fp32
    char* regX   = take((size_t)M0 * 184 * 2);             // enc3 out (bf16)
    char* hC     = take((size_t)8192 * 184 * 2);
    char* cC     = take((size_t)8192 * 180 * 4);
    char* y1L    = take((size_t)8192 * 200 * 2);
    char* leafO  = take((size_t)8192 * 184 * 2);
    char* gxN    = take((size_t)8192 * 720 * 2);
    char* hN     = take((size_t)4096 * 184 * 2);
    char* cN     = take((size_t)4096 * 180 * 4);
    char* y1I    = take((size_t)4096 * 200 * 2);
    char* intO   = take((size_t)4096 * 184 * 2);
    char* gxR    = take((size_t)4096 * 720 * 2);
    char* hR     = take((size_t)2048 * 184 * 2);
    char* cR     = take((size_t)2048 * 180 * 4);
    char* y1R    = take((size_t)2048 * 200 * 2);
    char* rootO  = take((size_t)2048 * 184 * 2);
    char* h1buf  = take((size_t)2048 * 352 * 2);
    char* h2f    = take((size_t)2048 * 180 * 4);
    unsigned short* wE1  = (unsigned short*)take((size_t)600 * 232 * 2);
    unsigned short* wE2  = (unsigned short*)take((size_t)350 * 600 * 2);
    unsigned short* wE3  = (unsigned short*)take((size_t)180 * 352 * 2);
    unsigned short* wH1  = (unsigned short*)take((size_t)350 * 184 * 2);
    unsigned short* wH2  = (unsigned short*)take((size_t)180 * 352 * 2);
    unsigned short* wC1L = (unsigned short*)take((size_t)200 * 184 * 2);
    unsigned short* wC1R = (unsigned short*)take((size_t)200 * 184 * 2);
    unsigned short* wC2  = (unsigned short*)take((size_t)180 * 200 * 2);
    unsigned short* wCih = (unsigned short*)take((size_t)720 * 184 * 2);
    unsigned short* wChh = (unsigned short*)take((size_t)720 * 184 * 2);
    unsigned short* wNih = (unsigned short*)take((size_t)720 * 184 * 2);
    unsigned short* wNhh = (unsigned short*)take((size_t)720 * 184 * 2);

    auto cvt = [&](const float* src, int srcld, int srcoff,
                   unsigned short* dst, int N, int K, int ldw) {
        int total = N * ldw;
        cvt_pad_kernel<<<(total + 255) / 256, 256, 0, stream>>>(
            src, srcld, srcoff, dst, N, K, ldw);
    };
    auto gemm = [&](const void* A, int aBf16, int lda,
                    const unsigned short* W, int ldw,
                    const float* bias, int act,
                    float* outF, unsigned short* outB, int ldc,
                    int M, int N, int K) {
        dim3 grid((N + BN - 1) / BN, (M + BM - 1) / BM);
        gemm_bf16_kernel<<<grid, 256, 0, stream>>>(
            A, aBf16, lda, W, ldw, bias, act, outF, outB, ldc, M, N, K);
    };

    // ---- 0. convert weights to padded bf16 ----
    cvt(W_e1, 228, 0,   wE1, 600, 228, 232);
    cvt(W_e2, 600, 0,   wE2, 350, 600, 600);
    cvt(W_e3, 350, 0,   wE3, 180, 350, 352);
    cvt(W_h1, 180, 0,   wH1, 350, 180, 184);
    cvt(W_h2, 350, 0,   wH2, 180, 350, 352);
    cvt(W_c1, 360, 0,   wC1L, 200, 180, 184);   // nodes_h half
    cvt(W_c1, 360, 180, wC1R, 200, 180, 184);   // comps_h half
    cvt(W_c2, 200, 0,   wC2, 180, 200, 200);
    cvt(comps_Wih, 180, 0, wCih, 720, 180, 184);
    cvt(comps_Whh, 180, 0, wChh, 720, 180, 184);
    cvt(nodes_Wih, 180, 0, wNih, 720, 180, 184);
    cvt(nodes_Whh, 180, 0, wNhh, 720, 180, 184);

    // ---- 1. encoder: x = elu(elu(elu(t@We1')@We2')@We3') ----
    gemm(tensor, 0, IN,  wE1, 232, nullptr, 1, nullptr, (unsigned short*)regA, 600, M0, 600, 228);
    gemm(regA,   1, 600, wE2, 600, nullptr, 1, nullptr, (unsigned short*)regB, 352, M0, 350, 600);
    gemm(regB,   1, 352, wE3, 352, nullptr, 1, nullptr, (unsigned short*)regX, 184, M0, 180, 352);

    // ---- 2. comps LSTM (4 leaves fused: batch 8192, 16 steps) ----
    gemm(regX, 1, 184, wCih, 184, nullptr, 0, nullptr, (unsigned short*)regA, 720, M0, 720, 184);
    hipMemsetAsync(hC, 0, (size_t)8192 * 184 * 2, stream);
    hipMemsetAsync(cC, 0, (size_t)8192 * 180 * 4, stream);
    float* Gstep = (float*)regB;
    for (int t = 0; t < 16; ++t) {
        gemm(hC, 1, 184, wChh, 184, nullptr, 0, Gstep, nullptr, 720, 8192, 720, 184);
        int total = 8192 * H;
        lstm_gate_kernel<<<(total + 255) / 256, 256, 0, stream>>>(
            Gstep, (const unsigned short*)regA, /*div*/4, /*outer*/64, /*inner*/16, /*off*/t,
            comps_bih, comps_bhh, (float*)cC, (unsigned short*)hC, 184, 8192, H);
    }

    // ---- 3. leaf MLP: y = elu(elu(comps_h @ Wc1R') @ Wc2') ----
    gemm(hC,  1, 184, wC1R, 184, nullptr, 1, nullptr, (unsigned short*)y1L, 200, 8192, 200, 184);
    gemm(y1L, 1, 200, wC2,  200, nullptr, 1, nullptr, (unsigned short*)leafO, 184, 8192, 180, 200);

    // ---- 4. internal-nodes LSTM (batch 4096, 2 steps over leaf outputs) ----
    gemm(leafO, 1, 184, wNih, 184, nullptr, 0, nullptr, (unsigned short*)gxN, 720, 8192, 720, 184);
    hipMemsetAsync(hN, 0, (size_t)4096 * 184 * 2, stream);
    hipMemsetAsync(cN, 0, (size_t)4096 * 180 * 4, stream);
    for (int t = 0; t < 2; ++t) {
        gemm(hN, 1, 184, wNhh, 184, nullptr, 0, Gstep, nullptr, 720, 4096, 720, 184);
        int total = 4096 * H;
        lstm_gate_kernel<<<(total + 255) / 256, 256, 0, stream>>>(
            Gstep, (const unsigned short*)gxN, /*div*/2, /*outer*/4, /*inner*/2, /*off*/t,
            nodes_bih, nodes_bhh, (float*)cN, (unsigned short*)hN, 184, 4096, H);
    }
    gemm(hN,  1, 184, wC1L, 184, nullptr, 1, nullptr, (unsigned short*)y1I, 200, 4096, 200, 184);
    gemm(y1I, 1, 200, wC2,  200, nullptr, 1, nullptr, (unsigned short*)intO, 184, 4096, 180, 200);

    // ---- 5. root LSTM (batch 2048, 2 steps over internal outputs) ----
    gemm(intO, 1, 184, wNih, 184, nullptr, 0, nullptr, (unsigned short*)gxR, 720, 4096, 720, 184);
    hipMemsetAsync(hR, 0, (size_t)2048 * 184 * 2, stream);
    hipMemsetAsync(cR, 0, (size_t)2048 * 180 * 4, stream);
    for (int t = 0; t < 2; ++t) {
        gemm(hR, 1, 184, wNhh, 184, nullptr, 0, Gstep, nullptr, 720, 2048, 720, 184);
        int total = 2048 * H;
        lstm_gate_kernel<<<(total + 255) / 256, 256, 0, stream>>>(
            Gstep, (const unsigned short*)gxR, /*div*/1, /*outer*/2, /*inner*/0, /*off*/t,
            nodes_bih, nodes_bhh, (float*)cR, (unsigned short*)hR, 184, 2048, H);
    }
    gemm(hR,  1, 184, wC1L, 184, nullptr, 1, nullptr, (unsigned short*)y1R, 200, 2048, 200, 184);
    gemm(y1R, 1, 200, wC2,  200, nullptr, 1, nullptr, (unsigned short*)rootO, 184, 2048, 180, 200);

    // ---- 6. head: elu(Wh1), elu(Wh2), dot(Wp)+bp, relu ----
    gemm(rootO, 1, 184, wH1, 184, nullptr, 1, nullptr, (unsigned short*)h1buf, 352, 2048, 350, 184);
    gemm(h1buf, 1, 352, wH2, 352, nullptr, 1, (float*)h2f, nullptr, 180, 2048, 180, 352);

    final_dot_kernel<<<(2048 * 32 + 255) / 256, 256, 0, stream>>>(
        (const float*)h2f, 180, W_p, b_p, out, 2048, H);
}
